// VQA_header_52931176956321
// MI455X (gfx1250) — matrix operand order repaired and checked by reading, not compiled, not run on hardware
//
#include <hip/hip_runtime.h>

typedef __attribute__((ext_vector_type(16))) _Float16 v16h;
typedef __attribute__((ext_vector_type(8)))  float    v8f;

#define B_ROWS   16384
#define D_INP    1024
#define D_HID    1000
#define D_HIDP   1024
#define MAX_ANS  3129
#define KT       32              // 32-wide K per WMMA, 32 total over K=1024
#define KSTAGE_H 128             // halves of K per pipeline stage
#define NSTAGE   (D_INP / KSTAGE_H)   // 8
#define ROW_B    272             // LDS bytes per row per stage (256 + 16 pad, conflict-free)
#define STAGE_B  (64 * ROW_B)    // 17408 B per stage buffer

union FragH { v16h h; uint4 q[2]; };

#define ASYNC_LD_B128(ldsoff, gaddr)                                        \
    asm volatile("global_load_async_to_lds_b128 %0, %1, off"                \
                 :: "v"(ldsoff), "v"(gaddr) : "memory")
#define WAIT_ASYNC0() asm volatile("s_wait_asynccnt 0" ::: "memory")

// ---------------- routing ----------------
__global__ void k_init_counts(int* counts) {
    if (threadIdx.x < 3) counts[threadIdx.x] = 0;
}

__global__ void k_route(const float* __restrict__ qt, int* __restrict__ counts,
                        int* __restrict__ idxL) {
    int r = blockIdx.x * 256 + threadIdx.x;
    if (r >= B_ROWS) return;
    float q0 = qt[3 * r], q1 = qt[3 * r + 1], q2 = qt[3 * r + 2];
    int e = 0; float bv = q0;
    if (q1 > bv) { e = 1; bv = q1; }
    if (q2 > bv) { e = 2; }
    int pos = atomicAdd(&counts[e], 1);
    idxL[e * B_ROWS + pos] = r;
}

// ---------------- conversion / packing ----------------
__global__ void k_cvt_x(const float4* __restrict__ x, uint2* __restrict__ xh, int n4) {
    int i = blockIdx.x * 256 + threadIdx.x;
    if (i >= n4) return;
    float4 v = x[i];
    union { _Float16 h[4]; uint2 u; } p;
    p.h[0] = (_Float16)v.x; p.h[1] = (_Float16)v.y;
    p.h[2] = (_Float16)v.z; p.h[3] = (_Float16)v.w;
    xh[i] = p.u;
}

// Pack [kRows x nCols] f32 row-major weights into WMMA B-operand tiles:
// tile index = ntG*KT + kt; within tile: lane-major, 16 halves per lane.
__global__ void k_pack_w(const float* __restrict__ src, _Float16* __restrict__ dst,
                         int kRows, int nCols, int nTilesN) {
    int p = blockIdx.x * 256 + threadIdx.x;
    int total = nTilesN * KT * 512;
    if (p >= total) return;
    int j    = p & 15;
    int lane = (p >> 4) & 31;
    int tile = p >> 9;
    int kt   = tile & (KT - 1);
    int ntG  = tile >> 5;
    int k = kt * 32 + ((j & 8) << 1) + ((lane & 16) >> 1) + (j & 7);
    int n = ntG * 16 + (lane & 15);
    float v = (k < kRows && n < nCols) ? src[k * nCols + n] : 0.0f;
    dst[p] = (_Float16)v;
}

__global__ void k_zero4(float4* __restrict__ out, int n4) {
    int i = blockIdx.x * 256 + threadIdx.x;
    if (i < n4) out[i] = make_float4(0.f, 0.f, 0.f, 0.f);
}

// ---------------- GEMM1: gathered X @ W1 -> tanh -> Hh (f16) ----------------
__global__ __launch_bounds__(256)
void k_gemm1(const _Float16* __restrict__ Xh, const _Float16* __restrict__ Wp,
             const float* __restrict__ b1, const int* __restrict__ cntPtr,
             const int* __restrict__ idxL, _Float16* __restrict__ Hh) {
    __shared__ int rows[64];
    __shared__ __align__(16) char As[2 * STAGE_B];
    int cnt = *cntPtr;
    int bm = blockIdx.x;
    if (bm * 64 >= cnt) return;
    int t = threadIdx.x;
    if (t < 64) {
        int g = bm * 64 + t;
        rows[t] = (g < cnt) ? idxL[g] : idxL[0];
    }
    __syncthreads();
    int lane = t & 31, wv = t >> 5;
    int mg = wv >> 1, ng = wv & 1;

    // async loader mapping: thread t handles rows r0+16k, k=0..3, 16B each
    int r0   = t >> 4;
    int coff = (t & 15) * 16;
    unsigned long long g64[4];
    unsigned           lo[4];
#pragma unroll
    for (int k = 0; k < 4; ++k) {
        int r = r0 + 16 * k;
        g64[k] = (unsigned long long)(uintptr_t)(Xh + (size_t)rows[r] * D_INP) + coff;
        lo[k]  = (unsigned)(r * ROW_B + coff);
    }
    unsigned lbase = (unsigned)(uintptr_t)(&As[0]);   // LDS byte offset (addr[31:0])
    const char* fb = As + (mg * 16 + (lane & 15)) * ROW_B + (lane & 16);

    auto loadB = [&](int kt, FragH b[4]) {
#pragma unroll
        for (int ai = 0; ai < 4; ++ai) {
            int ntG = blockIdx.y * 8 + ng * 4 + ai;
            const _Float16* bp = Wp + ((size_t)(ntG * KT + kt)) * 512 + lane * 16;
            b[ai].q[0] = *(const uint4*)(bp);
            b[ai].q[1] = *(const uint4*)(bp + 8);
        }
    };

    // prologue: stage 0 into buffer 0; B tiles for kt=0
#pragma unroll
    for (int k = 0; k < 4; ++k) ASYNC_LD_B128(lbase + lo[k], g64[k]);
    FragH bc[4], bn[4];
    loadB(0, bc);

    v8f acc[4] = {};
    for (int s = 0; s < NSTAGE; ++s) {
        WAIT_ASYNC0();
        __syncthreads();
        if (s + 1 < NSTAGE) {
            unsigned nb = (unsigned)((s + 1) & 1) * STAGE_B;
#pragma unroll
            for (int k = 0; k < 4; ++k)
                ASYNC_LD_B128(lbase + nb + lo[k], g64[k] + (unsigned long long)(s + 1) * 256);
        }
        const char* fbs = fb + (s & 1) * STAGE_B;
#pragma unroll
        for (int ksub = 0; ksub < 4; ++ksub) {
            int kt = s * 4 + ksub;
            if (kt + 1 < KT) loadB(kt + 1, bn);     // prefetch next B while computing
            FragH a;
            a.q[0] = *(const uint4*)(fbs + ksub * 64);
            a.q[1] = *(const uint4*)(fbs + ksub * 64 + 32);
#pragma unroll
            for (int ai = 0; ai < 4; ++ai)
                acc[ai] = __builtin_amdgcn_wmma_f32_16x16x32_f16(
                    false, a.h, false, bc[ai].h, (short)0, acc[ai], false, false);
#pragma unroll
            for (int ai = 0; ai < 4; ++ai) { bc[ai].q[0] = bn[ai].q[0]; bc[ai].q[1] = bn[ai].q[1]; }
        }
    }

    int mbase = bm * 64 + mg * 16 + ((lane >> 4) << 3);
    int nll = lane & 15;
#pragma unroll
    for (int ai = 0; ai < 4; ++ai) {
        int col = blockIdx.y * 128 + ng * 64 + ai * 16 + nll;
        float bias = (col < D_HID) ? b1[col] : 0.0f;
#pragma unroll
        for (int r = 0; r < 8; ++r) {
            int gm = mbase + r;
            if (gm < cnt)
                Hh[(size_t)gm * D_HIDP + col] = (_Float16)tanhf(acc[ai][r] + bias);
        }
    }
}

// ---------------- GEMM2: Hh @ W2 + b2 -> scatter to out ----------------
__global__ __launch_bounds__(256)
void k_gemm2(const _Float16* __restrict__ Hh, const _Float16* __restrict__ Wp,
             const float* __restrict__ b2, const int* __restrict__ cntPtr,
             const int* __restrict__ idxL, int nOut, float* __restrict__ out) {
    __shared__ int rows[64];
    __shared__ __align__(16) char As[2 * STAGE_B];
    int cnt = *cntPtr;
    int bm = blockIdx.x;
    if (bm * 64 >= cnt) return;
    int t = threadIdx.x;
    if (t < 64) {
        int g = bm * 64 + t;
        rows[t] = (g < cnt) ? idxL[g] : 0;
    }
    __syncthreads();
    int lane = t & 31, wv = t >> 5;
    int mg = wv >> 1, ng = wv & 1;

    int r0   = t >> 4;
    int coff = (t & 15) * 16;
    unsigned long long g64[4];
    unsigned           lo[4];
#pragma unroll
    for (int k = 0; k < 4; ++k) {
        int r = r0 + 16 * k;
        int gA = bm * 64 + r;
        if (gA >= cnt) gA = cnt - 1;
        g64[k] = (unsigned long long)(uintptr_t)(Hh + (size_t)gA * D_HIDP) + coff;
        lo[k]  = (unsigned)(r * ROW_B + coff);
    }
    unsigned lbase = (unsigned)(uintptr_t)(&As[0]);
    const char* fb = As + (mg * 16 + (lane & 15)) * ROW_B + (lane & 16);

    auto loadB = [&](int kt, FragH b[4]) {
#pragma unroll
        for (int ai = 0; ai < 4; ++ai) {
            int ntG = blockIdx.y * 8 + ng * 4 + ai;
            const _Float16* bp = Wp + ((size_t)(ntG * KT + kt)) * 512 + lane * 16;
            b[ai].q[0] = *(const uint4*)(bp);
            b[ai].q[1] = *(const uint4*)(bp + 8);
        }
    };

#pragma unroll
    for (int k = 0; k < 4; ++k) ASYNC_LD_B128(lbase + lo[k], g64[k]);
    FragH bc[4], bn[4];
    loadB(0, bc);

    v8f acc[4] = {};
    for (int s = 0; s < NSTAGE; ++s) {
        WAIT_ASYNC0();
        __syncthreads();
        if (s + 1 < NSTAGE) {
            unsigned nb = (unsigned)((s + 1) & 1) * STAGE_B;
#pragma unroll
            for (int k = 0; k < 4; ++k)
                ASYNC_LD_B128(lbase + nb + lo[k], g64[k] + (unsigned long long)(s + 1) * 256);
        }
        const char* fbs = fb + (s & 1) * STAGE_B;
#pragma unroll
        for (int ksub = 0; ksub < 4; ++ksub) {
            int kt = s * 4 + ksub;
            if (kt + 1 < KT) loadB(kt + 1, bn);
            FragH a;
            a.q[0] = *(const uint4*)(fbs + ksub * 64);
            a.q[1] = *(const uint4*)(fbs + ksub * 64 + 32);
#pragma unroll
            for (int ai = 0; ai < 4; ++ai)
                acc[ai] = __builtin_amdgcn_wmma_f32_16x16x32_f16(
                    false, a.h, false, bc[ai].h, (short)0, acc[ai], false, false);
#pragma unroll
            for (int ai = 0; ai < 4; ++ai) { bc[ai].q[0] = bn[ai].q[0]; bc[ai].q[1] = bn[ai].q[1]; }
        }
    }

    int mloc = mg * 16 + ((lane >> 4) << 3);
    int nll = lane & 15;
#pragma unroll
    for (int ai = 0; ai < 4; ++ai) {
        int col = blockIdx.y * 128 + ng * 64 + ai * 16 + nll;
        if (col < nOut) {
            float bias = b2[col];
#pragma unroll
            for (int r = 0; r < 8; ++r) {
                int gm = bm * 64 + mloc + r;
                if (gm < cnt)
                    out[(size_t)rows[mloc + r] * MAX_ANS + col] = acc[ai][r] + bias;
            }
        }
    }
}

// ---------------- launch ----------------
extern "C" void kernel_launch(void* const* d_in, const int* in_sizes, int n_in,
                              void* d_out, int out_size, void* d_ws, size_t ws_size,
                              hipStream_t stream) {
    (void)in_sizes; (void)n_in; (void)out_size; (void)ws_size;
    const float* hs = (const float*)d_in[0];
    const float* qt = (const float*)d_in[1];
    const float* w1[3] = { (const float*)d_in[2], (const float*)d_in[6], (const float*)d_in[10] };
    const float* b1[3] = { (const float*)d_in[3], (const float*)d_in[7], (const float*)d_in[11] };
    const float* w2[3] = { (const float*)d_in[4], (const float*)d_in[8], (const float*)d_in[12] };
    const float* b2[3] = { (const float*)d_in[5], (const float*)d_in[9], (const float*)d_in[13] };
    float* out = (float*)d_out;

    const int nOut[3]  = { 2, 482, 3129 };
    const int nPad[3]  = { 128, 512, 3200 };   // N padded to 128 (WG N-tile)

    char* ws = (char*)d_ws;
    int* counts = (int*)ws;                                    // 16 B
    int* idxL   = (int*)(ws + 256);                            // 3 * 16384 * 4
    _Float16* Xh  = (_Float16*)(ws + 262144);                  // 16384*1024 f16
    _Float16* Hh  = Xh + (size_t)B_ROWS * D_INP;               // 16384*1024 f16
    _Float16* W1h = Hh + (size_t)B_ROWS * D_HIDP;              // 3 * 1024*1024 f16
    _Float16* W2h[3];
    W2h[0] = W1h + 3ull * 1024 * 1024;
    W2h[1] = W2h[0] + (size_t)nPad[0] * 1024;
    W2h[2] = W2h[1] + (size_t)nPad[1] * 1024;

    // 1) routing
    k_init_counts<<<1, 32, 0, stream>>>(counts);
    k_route<<<B_ROWS / 256, 256, 0, stream>>>(qt, counts, idxL);

    // 2) convert X, pack weights
    int nx4 = B_ROWS * D_INP / 4;
    k_cvt_x<<<nx4 / 256, 256, 0, stream>>>((const float4*)hs, (uint2*)Xh, nx4);
    for (int e = 0; e < 3; ++e) {
        int ntg1 = D_HIDP / 16;  // 64
        k_pack_w<<<(ntg1 * KT * 512) / 256, 256, 0, stream>>>(
            w1[e], W1h + (size_t)e * 1024 * 1024, D_INP, D_HID, ntg1);
        int ntg2 = nPad[e] / 16;
        k_pack_w<<<(ntg2 * KT * 512) / 256, 256, 0, stream>>>(
            w2[e], W2h[e], D_HID, nOut[e], ntg2);
    }

    // 3) zero output (padding + poison cleanup)
    int no4 = (B_ROWS * MAX_ANS) / 4;
    k_zero4<<<(no4 + 255) / 256, 256, 0, stream>>>((float4*)out, no4);

    // 4) per-expert pipelined GEMMs (fixed grids, early-exit on count)
    for (int e = 0; e < 3; ++e) {
        dim3 g1(B_ROWS / 64, D_HIDP / 128);
        k_gemm1<<<g1, 256, 0, stream>>>(Xh, W1h + (size_t)e * 1024 * 1024, b1[e],
                                        counts + e, idxL + e * B_ROWS, Hh);
        dim3 g2(B_ROWS / 64, nPad[e] / 128);
        k_gemm2<<<g2, 256, 0, stream>>>(Hh, W2h[e], b2[e],
                                        counts + e, idxL + e * B_ROWS, nOut[e], out);
    }
}
